// PTA_EGNN_32933809225924
// MI455X (gfx1250) — matrix operand order repaired and checked by reading, not compiled, not run on hardware
//
#include <hip/hip_runtime.h>

typedef __attribute__((ext_vector_type(16))) _Float16 v16h;
typedef __attribute__((ext_vector_type(8)))  _Float16 v8h;
typedef __attribute__((ext_vector_type(2)))  _Float16 v2h;
typedef __attribute__((ext_vector_type(8)))  float    v8f;

constexpr int NN   = 10000;     // nodes
constexpr int NE   = 200000;    // edges
constexpr int DD   = 64;        // feature dim
constexpr int EDIM = 16;        // edge_attr dim
constexpr int CIN  = 4*DD + EDIM;   // 272
constexpr int CINP = 288;           // padded to multiple of 32

__device__ __forceinline__ float siluf(float x){ return x * (1.0f/(1.0f+__expf(-x))); }
__device__ __forceinline__ float sigmf(float x){ return 1.0f/(1.0f+__expf(-x)); }
__device__ __forceinline__ int   imin(int a,int b){ return a<b?a:b; }

// ---- WMMA fragment helpers (layouts per CDNA5 ISA 7.12.2, wave32) ----
// A: 16xK f16 tile in LDS, row-major, 16B-aligned rows. Lane l, half j:
//   row = l%16, k = kk + 8*(l/16) + (j<8 ? j : j+8)
// -> two contiguous 16B chunks per lane -> 2x ds_load_b128.
__device__ __forceinline__ v16h load_a_lds(const _Float16* A, int lda, int kk, int lane){
  const _Float16* p = A + (lane & 15)*lda + kk + ((lane >> 4) << 3);
  v8h lo = *(const v8h*)p;
  v8h hh = *(const v8h*)(p + 16);
  return __builtin_shufflevector(lo, hh, 0,1,2,3,4,5,6,7,8,9,10,11,12,13,14,15);
}
// B: weights pre-packed in fragment order: 32B contiguous per (kb,cb,lane).
__device__ __forceinline__ v16h load_b_pk(const _Float16* Wp, int kb, int wave, int lane){
  return *(const v16h*)(Wp + ((((size_t)kb*4 + wave)*32 + lane) << 4));
}
// 16x64 = (16xK) * (Kx64); each of 4 waves produces its 16-col block.
template<int K>
__device__ __forceinline__ v8f gemm_tile(const _Float16* A, int lda, const _Float16* Wp,
                                         int wave, int lane){
  v8f acc = {};
  #pragma unroll
  for (int kb = 0; kb < K/32; ++kb){
    v16h a = load_a_lds(A, lda, kb*32, lane);
    v16h b = load_b_pk(Wp, kb, wave, lane);
    acc = __builtin_amdgcn_wmma_f32_16x16x32_f16(false, a, false, b, (short)0, acc, false, false);
  }
  return acc;
}

// ---- pack f32 KxN(=64) weights into f16 WMMA-B fragment order (zero pad K) ----
// dst[o]: o = (((kb*4 + cb)*32 + lane)*16 + j), element = W[kb*32 + 8*(lane/16) + (j<8 ? j : j+8)][cb*16 + lane%16]
__global__ void pack_b_kernel(const float* __restrict__ src, _Float16* __restrict__ dst,
                              int srcK, int dstK){
  int o = blockIdx.x*256 + threadIdx.x;
  if (o >= dstK*DD) return;
  int j  = o & 15;
  int l  = (o >> 4) & 31;
  int cb = (o >> 9) & 3;
  int kb = o >> 11;
  int k  = kb*32 + ((l >> 4) << 3) + (j < 8 ? j : j + 8);
  int col= cb*16 + (l & 15);
  float v = (k < srcK) ? src[(size_t)k*DD + col] : 0.0f;
  dst[o] = (_Float16)v;
}

// ---- Kernel 1: per-edge geometry + chem MLP + pos MLP + gate z + phi_u/x/n ----
__global__ void __launch_bounds__(128)
edge_kernel(const float* __restrict__ h, const float* __restrict__ coord,
            const float* __restrict__ nvecs, const float* __restrict__ edge_attr,
            const float* __restrict__ node_attr, const int* __restrict__ edges,
            const float* __restrict__ ch_b1, const float* __restrict__ ch_b2,
            const float* __restrict__ pos_W1, const float* __restrict__ pos_b1,
            const float* __restrict__ pos_b2,
            const float* __restrict__ sh_b, const float* __restrict__ att_W,
            const float* __restrict__ att_b,
            const float* __restrict__ phiu_b1, const float* __restrict__ phiu_W2,
            const float* __restrict__ phiu_b2,
            const float* __restrict__ phix_b1, const float* __restrict__ phix_W2,
            const float* __restrict__ phix_b2,
            const float* __restrict__ phin_b1, const float* __restrict__ phin_W2,
            const float* __restrict__ phin_b2,
            const _Float16* __restrict__ W1p, const _Float16* __restrict__ W2p,
            const _Float16* __restrict__ posW2p, const _Float16* __restrict__ shWp,
            const _Float16* __restrict__ phiuW1p, const _Float16* __restrict__ phixW1p,
            const _Float16* __restrict__ phinW1p,
            _Float16* __restrict__ zf16, float* __restrict__ pupx,
            float* __restrict__ pupn, float* __restrict__ cdiff)
{
  __shared__ alignas(16) _Float16 sIn[16*CINP];
  __shared__ alignas(16) _Float16 sA[16*DD];
  __shared__ alignas(16) _Float16 sChem[16*DD];
  __shared__ alignas(16) _Float16 sPosH[16*DD];
  __shared__ float    sPos[16*DD];
  __shared__ float    sF[16*DD];
  __shared__ float    sScal[32];
  __shared__ float    sPU[16];
  __shared__ int      sRC[32];

  const int e0   = blockIdx.x * 16;
  const int t    = threadIdx.x;
  const int lane = t & 31;
  const int wave = t >> 5;

  // 1. per-edge geometry
  if (t < 16){
    int e = imin(e0 + t, NE-1);
    int r = edges[e], c = edges[NE + e];
    sRC[t*2] = r; sRC[t*2+1] = c;
    float dx = coord[r*3+0]-coord[c*3+0];
    float dy = coord[r*3+1]-coord[c*3+1];
    float dz = coord[r*3+2]-coord[c*3+2];
    float rad = dx*dx+dy*dy+dz*dz;
    float inv = 1.0f/(sqrtf(rad)+1e-8f);
    float np  = nvecs[r*3+0]*nvecs[c*3+0]+nvecs[r*3+1]*nvecs[c*3+1]+nvecs[r*3+2]*nvecs[c*3+2];
    sScal[t*2] = np; sScal[t*2+1] = rad;
    if (e0 + t < NE){
      cdiff[(size_t)e*3+0] = dx*inv; cdiff[(size_t)e*3+1] = dy*inv; cdiff[(size_t)e*3+2] = dz*inv;
    }
  }
  __syncthreads();

  // 2. build chem_in tile (16 x 288 f16)
  for (int idx = t; idx < 16*CINP; idx += 128){
    int i = idx / CINP, f = idx % CINP;
    int e = imin(e0 + i, NE-1);
    int r = sRC[i*2], c = sRC[i*2+1];
    float v;
    if      (f < 64)  v = h[(size_t)r*DD + f];
    else if (f < 128) v = h[(size_t)c*DD + (f-64)];
    else if (f < 192) v = node_attr[(size_t)r*DD + (f-128)];
    else if (f < 256) v = node_attr[(size_t)c*DD + (f-192)];
    else if (f < CIN) v = edge_attr[(size_t)e*EDIM + (f-256)];
    else              v = 0.0f;
    sIn[i*CINP + f] = (_Float16)v;
  }
  __syncthreads();

  // 3. chem layer 1 (K=288)
  {
    v8f acc = gemm_tile<CINP>(sIn, CINP, W1p, wave, lane);
    int col = wave*16 + (lane & 15), hi = lane >> 4;
    #pragma unroll
    for (int i = 0; i < 8; ++i)
      sA[(i + 8*hi)*DD + col] = (_Float16)siluf(acc[i] + ch_b1[col]);
  }
  __syncthreads();

  // 4. chem layer 2 (K=64)
  {
    v8f acc = gemm_tile<DD>(sA, DD, W2p, wave, lane);
    int col = wave*16 + (lane & 15), hi = lane >> 4;
    #pragma unroll
    for (int i = 0; i < 8; ++i)
      sChem[(i + 8*hi)*DD + col] = (_Float16)siluf(acc[i] + ch_b2[col]);
  }
  __syncthreads();

  // 5. pos layer 1 (2 -> 64, VALU)
  for (int idx = t; idx < 16*DD; idx += 128){
    int i = idx >> 6, j = idx & 63;
    float p = siluf(sScal[i*2]*pos_W1[j] + sScal[i*2+1]*pos_W1[DD+j] + pos_b1[j]);
    sA[idx] = (_Float16)p;
  }
  __syncthreads();

  // 6. pos layer 2 (K=64)
  {
    v8f acc = gemm_tile<DD>(sA, DD, posW2p, wave, lane);
    int col = wave*16 + (lane & 15), hi = lane >> 4;
    #pragma unroll
    for (int i = 0; i < 8; ++i){
      float p = siluf(acc[i] + pos_b2[col]);
      sPos [(i + 8*hi)*DD + col] = p;
      sPosH[(i + 8*hi)*DD + col] = (_Float16)p;
    }
  }
  __syncthreads();

  // 7. out = silu(chem @ sh_W + sh_b) * pos
  {
    v8f acc = gemm_tile<DD>(sChem, DD, shWp, wave, lane);
    int col = wave*16 + (lane & 15), hi = lane >> 4;
    #pragma unroll
    for (int i = 0; i < 8; ++i){
      int r = i + 8*hi;
      sF[r*DD + col] = siluf(acc[i] + sh_b[col]) * sPos[r*DD + col];
    }
  }
  __syncthreads();

  // 8. attention gate z = out * sigmoid(out @ att_W + att_b)
  if (t < 16){
    float s = att_b[0];
    for (int j = 0; j < DD; ++j) s += sF[t*DD + j]*att_W[j];
    sScal[t*2] = sigmf(s);
  }
  __syncthreads();
  {
    // 128 threads x 8 contiguous halves = 16x64 tile, 16B vector stores
    int i = t >> 3, j0 = (t & 7)*8;
    if (e0 + i < NE){
      v8h zv;
      #pragma unroll
      for (int jj = 0; jj < 8; ++jj) zv[jj] = (_Float16)(sF[i*DD + j0 + jj]*sScal[i*2]);
      *(v8h*)(zf16 + (size_t)(e0+i)*DD + j0) = zv;
    }
  }
  __syncthreads();

  // 9. phi_u = mlp2(chem): silu layer via WMMA, then 64->1 dot
  {
    v8f acc = gemm_tile<DD>(sChem, DD, phiuW1p, wave, lane);
    int col = wave*16 + (lane & 15), hi = lane >> 4;
    #pragma unroll
    for (int i = 0; i < 8; ++i)
      sF[(i + 8*hi)*DD + col] = siluf(acc[i] + phiu_b1[col]);
  }
  __syncthreads();
  if (t < 16){
    float s = phiu_b2[0];
    for (int j = 0; j < DD; ++j) s += sF[t*DD + j]*phiu_W2[j];
    sPU[t] = s;
  }
  __syncthreads();

  // 10. phi_x = mlp2(pos)
  {
    v8f acc = gemm_tile<DD>(sPosH, DD, phixW1p, wave, lane);
    int col = wave*16 + (lane & 15), hi = lane >> 4;
    #pragma unroll
    for (int i = 0; i < 8; ++i)
      sF[(i + 8*hi)*DD + col] = siluf(acc[i] + phix_b1[col]);
  }
  __syncthreads();
  if (t < 16){
    float s = phix_b2[0];
    for (int j = 0; j < DD; ++j) s += sF[t*DD + j]*phix_W2[j];
    if (e0 + t < NE) pupx[e0 + t] = sPU[t]*s;
  }
  __syncthreads();

  // 11. phi_n = mlp2(pos)
  {
    v8f acc = gemm_tile<DD>(sPosH, DD, phinW1p, wave, lane);
    int col = wave*16 + (lane & 15), hi = lane >> 4;
    #pragma unroll
    for (int i = 0; i < 8; ++i)
      sF[(i + 8*hi)*DD + col] = siluf(acc[i] + phin_b1[col]);
  }
  __syncthreads();
  if (t < 16){
    float s = phin_b2[0];
    for (int j = 0; j < DD; ++j) s += sF[t*DD + j]*phin_W2[j];
    if (e0 + t < NE) pupn[e0 + t] = sPU[t]*s;
  }
}

// ---- Kernel 2: per-head q,k,v,g (WMMA) + b (dot), from z ----
__global__ void __launch_bounds__(128)
qkvg_kernel(const _Float16* __restrict__ zf16,
            const _Float16* __restrict__ Wqp, const _Float16* __restrict__ Wkp,
            const _Float16* __restrict__ Wvp, const _Float16* __restrict__ Wgp,
            const float* __restrict__ Wb, const float* __restrict__ bg, int head,
            _Float16* __restrict__ q, _Float16* __restrict__ k,
            _Float16* __restrict__ v, _Float16* __restrict__ g,
            float* __restrict__ b)
{
  __shared__ alignas(16) _Float16 sZ[16*DD];
  const int e0 = blockIdx.x*16;
  const int t = threadIdx.x, lane = t & 31, wave = t >> 5;

  {
    int i = t >> 3, j0 = (t & 7)*8;
    int e = imin(e0 + i, NE-1);
    *(v8h*)(sZ + i*DD + j0) = *(const v8h*)(zf16 + (size_t)e*DD + j0);
  }
  __syncthreads();

  const size_t hw = (size_t)head*DD*DD;   // packed head stride = 64*64 elems
  int col = wave*16 + (lane & 15), hi = lane >> 4;

  {
    v8f acc = gemm_tile<DD>(sZ, DD, Wqp + hw, wave, lane);
    #pragma unroll
    for (int i = 0; i < 8; ++i){ int e = e0 + i + 8*hi;
      if (e < NE) q[(size_t)e*DD + col] = (_Float16)acc[i]; }
  }
  {
    v8f acc = gemm_tile<DD>(sZ, DD, Wkp + hw, wave, lane);
    #pragma unroll
    for (int i = 0; i < 8; ++i){ int e = e0 + i + 8*hi;
      if (e < NE) k[(size_t)e*DD + col] = (_Float16)acc[i]; }
  }
  {
    v8f acc = gemm_tile<DD>(sZ, DD, Wvp + hw, wave, lane);
    #pragma unroll
    for (int i = 0; i < 8; ++i){ int e = e0 + i + 8*hi;
      if (e < NE) v[(size_t)e*DD + col] = (_Float16)acc[i]; }
  }
  {
    v8f acc = gemm_tile<DD>(sZ, DD, Wgp + hw, wave, lane);
    #pragma unroll
    for (int i = 0; i < 8; ++i){ int e = e0 + i + 8*hi;
      if (e < NE) g[(size_t)e*DD + col] = (_Float16)sigmf(acc[i] + bg[head*DD + col]); }
  }
  if (t < 16){
    float s = 0.0f;
    for (int j = 0; j < DD; ++j) s += (float)sZ[t*DD + j]*Wb[head*DD + j];
    if (e0 + t < NE) b[e0 + t] = s;
  }
}

// ---- Kernel 3: attention over K=16 gathered neighbors (wave per edge) ----
__global__ void __launch_bounds__(128)
attn_kernel(const _Float16* __restrict__ q, const _Float16* __restrict__ k,
            const _Float16* __restrict__ v, const _Float16* __restrict__ g,
            const float* __restrict__ b, const int* __restrict__ klist,
            _Float16* __restrict__ ho)
{
  __shared__ float sAl[4][16];
  const int lane = threadIdx.x & 31;
  const int wave = threadIdx.x >> 5;
  const int e    = blockIdx.x*4 + wave;
  const bool valid = (e < NE);
  const int ec   = valid ? e : 0;
  const int* ki  = klist + (size_t)ec*32;   // [e][0][*] then [e][1][*]

  if (lane < 16){
    float score = 0.0f;
    int idx = ki[lane];
    if (idx >= 0){
      __builtin_prefetch(v + (size_t)idx*DD, 0, 1);   // warm the v-row (global_prefetch_b8)
      float s = 0.0f;
      const v8h* qp = (const v8h*)(q + (size_t)ec*DD);
      const v8h* kp = (const v8h*)(k + (size_t)idx*DD);
      #pragma unroll
      for (int c = 0; c < 8; ++c){
        v8h qa = qp[c], ka = kp[c];
        #pragma unroll
        for (int jj = 0; jj < 8; ++jj) s += (float)qa[jj]*(float)ka[jj];
      }
      int bj = ki[16 + lane];
      score = 0.125f*s + ((bj >= 0) ? b[bj] : 0.0f);  // scale = 1/sqrt(64)
    }
    sAl[wave][lane] = score;
  }
  __syncthreads();

  float mx = -1e30f;
  #pragma unroll
  for (int n = 0; n < 16; ++n) mx = fmaxf(mx, sAl[wave][n]);
  float al[16]; float sm = 0.0f;
  #pragma unroll
  for (int n = 0; n < 16; ++n){ al[n] = __expf(sAl[wave][n]-mx); sm += al[n]; }
  const float inv = 1.0f/sm;

  // lane handles 2 contiguous dims -> 4B gathers / stores
  float o0 = 0.0f, o1 = 0.0f;
  #pragma unroll
  for (int n = 0; n < 16; ++n){
    int idx = ki[n];
    if (idx >= 0){
      float w = al[n]*inv;
      v2h vv = *(const v2h*)(v + (size_t)idx*DD + lane*2);
      o0 += w*(float)vv[0];
      o1 += w*(float)vv[1];
    }
  }
  v2h gg = *(const v2h*)(g + (size_t)ec*DD + lane*2);
  if (valid){
    v2h hv; hv[0] = (_Float16)(o0*(float)gg[0]); hv[1] = (_Float16)(o1*(float)gg[1]);
    *(v2h*)(ho + (size_t)e*DD + lane*2) = hv;
  }
}

// ---- Kernel 4: m = concat(ho0..ho3) @ out_W + out_b  (K=256 WMMA) ----
__global__ void __launch_bounds__(128)
m_kernel(const _Float16* __restrict__ ho0, const _Float16* __restrict__ ho1,
         const _Float16* __restrict__ ho2, const _Float16* __restrict__ ho3,
         const _Float16* __restrict__ outWp, const float* __restrict__ out_b,
         float* __restrict__ m)
{
  __shared__ alignas(16) _Float16 sIn[16*256];
  const int e0 = blockIdx.x*16;
  const int t = threadIdx.x, lane = t & 31, wave = t >> 5;

  for (int c = t; c < 512; c += 128){          // 512 chunks of 8 halves
    int i = c >> 5, f0 = (c & 31)*8;
    int e = imin(e0 + i, NE-1);
    const _Float16* src = (f0 < 64) ? ho0 : (f0 < 128) ? ho1 : (f0 < 192) ? ho2 : ho3;
    *(v8h*)(sIn + i*256 + f0) = *(const v8h*)(src + (size_t)e*DD + (f0 & 63));
  }
  __syncthreads();

  v8f acc = gemm_tile<256>(sIn, 256, outWp, wave, lane);
  int col = wave*16 + (lane & 15), hi = lane >> 4;
  #pragma unroll
  for (int i = 0; i < 8; ++i){
    int e = e0 + i + 8*hi;
    if (e < NE) m[(size_t)e*DD + col] = acc[i] + out_b[col];
  }
}

// ---- Kernel 5a: zero accumulators ----
__global__ void zero_kernel(float* __restrict__ p, int n){
  int i = blockIdx.x*256 + threadIdx.x;
  if (i < n) p[i] = 0.0f;
}

// ---- Kernel 5b: atomic scatter (segment sums + counts) ----
__global__ void scatter_kernel(const float* __restrict__ m, const float* __restrict__ pupx,
                               const float* __restrict__ pupn, const float* __restrict__ cdiff,
                               const float* __restrict__ nvecs, const int* __restrict__ edges,
                               float* __restrict__ xsum, float* __restrict__ nsum,
                               float* __restrict__ msum, float* __restrict__ cnt)
{
  int gid = blockIdx.x*256 + threadIdx.x;
  if (gid >= NE*DD) return;
  int e = gid >> 6, d = gid & 63;
  int r = edges[e];
  atomicAdd(&msum[(size_t)r*DD + d], m[gid]);
  if (d == 0){
    int c = edges[NE + e];
    float ux = pupx[e], un = pupn[e];
    atomicAdd(&xsum[r*3+0], cdiff[(size_t)e*3+0]*ux);
    atomicAdd(&xsum[r*3+1], cdiff[(size_t)e*3+1]*ux);
    atomicAdd(&xsum[r*3+2], cdiff[(size_t)e*3+2]*ux);
    atomicAdd(&nsum[r*3+0], nvecs[c*3+0]*un);
    atomicAdd(&nsum[r*3+1], nvecs[c*3+1]*un);
    atomicAdd(&nsum[r*3+2], nvecs[c*3+2]*un);
    atomicAdd(&cnt[r], 1.0f);
  }
}

// ---- Kernel 6: node update (phih MLP via WMMA + coord/nvec epilogue) ----
__global__ void __launch_bounds__(128)
node_kernel(const float* __restrict__ h, const float* __restrict__ node_attr,
            const float* __restrict__ coord, const float* __restrict__ nvecs,
            const float* __restrict__ init_coord, const float* __restrict__ init_nvecs,
            const float* __restrict__ xsum, const float* __restrict__ nsum,
            const float* __restrict__ msum, const float* __restrict__ cnt,
            const _Float16* __restrict__ phihW1p, const float* __restrict__ phih_b1,
            const _Float16* __restrict__ phihW2p, const float* __restrict__ phih_b2,
            float* __restrict__ out)
{
  __shared__ alignas(16) _Float16 sIn[16*192];
  __shared__ alignas(16) _Float16 sA[16*DD];
  const int n0 = blockIdx.x*16;
  const int t = threadIdx.x, lane = t & 31, wave = t >> 5;

  for (int idx = t; idx < 16*192; idx += 128){
    int i = idx / 192, f = idx % 192;
    int n = imin(n0 + i, NN-1);
    float val;
    if      (f < 64)  val = h[(size_t)n*DD + f];
    else if (f < 128) val = node_attr[(size_t)n*DD + (f-64)];
    else              val = msum[(size_t)n*DD + (f-128)] / fmaxf(cnt[n], 1.0f);
    sIn[idx] = (_Float16)val;
  }
  __syncthreads();

  {
    v8f acc = gemm_tile<192>(sIn, 192, phihW1p, wave, lane);
    int col = wave*16 + (lane & 15), hi = lane >> 4;
    #pragma unroll
    for (int i = 0; i < 8; ++i)
      sA[(i + 8*hi)*DD + col] = (_Float16)siluf(acc[i] + phih_b1[col]);
  }
  __syncthreads();
  {
    v8f acc = gemm_tile<DD>(sA, DD, phihW2p, wave, lane);
    int col = wave*16 + (lane & 15), hi = lane >> 4;
    #pragma unroll
    for (int i = 0; i < 8; ++i){
      int n = n0 + i + 8*hi;
      if (n < NN){
        float val = acc[i] + phih_b2[col];
        out[(size_t)n*DD + col] = 0.2f*h[(size_t)n*DD + col] + 0.8f*val;
      }
    }
  }

  if (t < 16){
    int n = n0 + t;
    if (n < NN){
      float c = fmaxf(cnt[n], 1.0f);
      float* co = out + (size_t)NN*DD;
      float* no = out + (size_t)NN*DD + (size_t)NN*3;
      float nvv[3];
      #pragma unroll
      for (int ax = 0; ax < 3; ++ax){
        co[n*3+ax] = 0.2f*init_coord[n*3+ax] + 0.8f*coord[n*3+ax] + xsum[n*3+ax]/c;
        nvv[ax]    = 0.2f*init_nvecs[n*3+ax] + 0.8f*nvecs[n*3+ax] + nsum[n*3+ax]/c;
      }
      float nrm = sqrtf(nvv[0]*nvv[0]+nvv[1]*nvv[1]+nvv[2]*nvv[2]) + 1e-8f;
      #pragma unroll
      for (int ax = 0; ax < 3; ++ax) no[n*3+ax] = nvv[ax]/nrm;
    }
  }
}

// ================== host launch ==================
extern "C" void kernel_launch(void* const* d_in, const int* in_sizes, int n_in,
                              void* d_out, int out_size, void* d_ws, size_t ws_size,
                              hipStream_t stream)
{
  (void)in_sizes; (void)n_in; (void)out_size; (void)ws_size;
  const float* h          = (const float*)d_in[0];
  const float* coord      = (const float*)d_in[1];
  const float* nvecs      = (const float*)d_in[2];
  const float* edge_attr  = (const float*)d_in[3];
  const float* node_attr  = (const float*)d_in[4];
  const float* init_coord = (const float*)d_in[5];
  const float* init_nvecs = (const float*)d_in[6];
  const float* ch_W1      = (const float*)d_in[7];
  const float* ch_b1      = (const float*)d_in[8];
  const float* ch_W2      = (const float*)d_in[9];
  const float* ch_b2      = (const float*)d_in[10];
  const float* pos_W1     = (const float*)d_in[11];
  const float* pos_b1     = (const float*)d_in[12];
  const float* pos_W2     = (const float*)d_in[13];
  const float* pos_b2     = (const float*)d_in[14];
  const float* sh_W       = (const float*)d_in[15];
  const float* sh_b       = (const float*)d_in[16];
  const float* att_W      = (const float*)d_in[17];
  const float* att_b      = (const float*)d_in[18];
  const float* Wq         = (const float*)d_in[19];
  const float* Wk         = (const float*)d_in[20];
  const float* Wv         = (const float*)d_in[21];
  const float* Wb         = (const float*)d_in[22];
  const float* Wg         = (const float*)d_in[23];
  const float* bg         = (const float*)d_in[24];
  const float* out_W      = (const float*)d_in[25];
  const float* out_b      = (const float*)d_in[26];
  const float* phiu_W1    = (const float*)d_in[27];
  const float* phiu_b1    = (const float*)d_in[28];
  const float* phiu_W2    = (const float*)d_in[29];
  const float* phiu_b2    = (const float*)d_in[30];
  const float* phix_W1    = (const float*)d_in[31];
  const float* phix_b1    = (const float*)d_in[32];
  const float* phix_W2    = (const float*)d_in[33];
  const float* phix_b2    = (const float*)d_in[34];
  const float* phin_W1    = (const float*)d_in[35];
  const float* phin_b1    = (const float*)d_in[36];
  const float* phin_W2    = (const float*)d_in[37];
  const float* phin_b2    = (const float*)d_in[38];
  const float* phih_W1    = (const float*)d_in[39];
  const float* phih_b1    = (const float*)d_in[40];
  const float* phih_W2    = (const float*)d_in[41];
  const float* phih_b2    = (const float*)d_in[42];
  const int*   edges      = (const int*)d_in[43];
  const int*   klist      = (const int*)d_in[44];
  float* out = (float*)d_out;

  // ---- workspace bump allocator (256B aligned) ----
  char* base = (char*)d_ws;
  size_t off = 0;
  auto carve = [&](size_t bytes) -> char* {
    char* p = base + off;
    off += (bytes + 255) & ~(size_t)255;
    return p;
  };
  _Float16* W1p     = (_Float16*)carve((size_t)CINP*DD*2);
  _Float16* W2p     = (_Float16*)carve((size_t)DD*DD*2);
  _Float16* posW2p  = (_Float16*)carve((size_t)DD*DD*2);
  _Float16* shWp    = (_Float16*)carve((size_t)DD*DD*2);
  _Float16* phiuW1p = (_Float16*)carve((size_t)DD*DD*2);
  _Float16* phixW1p = (_Float16*)carve((size_t)DD*DD*2);
  _Float16* phinW1p = (_Float16*)carve((size_t)DD*DD*2);
  _Float16* Wqp     = (_Float16*)carve((size_t)4*DD*DD*2);
  _Float16* Wkp     = (_Float16*)carve((size_t)4*DD*DD*2);
  _Float16* Wvp     = (_Float16*)carve((size_t)4*DD*DD*2);
  _Float16* Wgp     = (_Float16*)carve((size_t)4*DD*DD*2);
  _Float16* outWp   = (_Float16*)carve((size_t)256*DD*2);
  _Float16* phihW1p = (_Float16*)carve((size_t)192*DD*2);
  _Float16* phihW2p = (_Float16*)carve((size_t)DD*DD*2);
  _Float16* zf16    = (_Float16*)carve((size_t)NE*DD*2);
  _Float16* qh      = (_Float16*)carve((size_t)NE*DD*2);
  _Float16* kh      = (_Float16*)carve((size_t)NE*DD*2);
  _Float16* vh      = (_Float16*)carve((size_t)NE*DD*2);
  _Float16* gh      = (_Float16*)carve((size_t)NE*DD*2);
  float*    bbuf    = (float*)   carve((size_t)NE*4);
  _Float16* ho[4];
  for (int i = 0; i < 4; ++i) ho[i] = (_Float16*)carve((size_t)NE*DD*2);
  float*    mbuf    = (float*)   carve((size_t)NE*DD*4);
  float*    pupx    = (float*)   carve((size_t)NE*4);
  float*    pupn    = (float*)   carve((size_t)NE*4);
  float*    cdiffb  = (float*)   carve((size_t)NE*3*4);
  const int accN    = NN*3 + NN*3 + NN*DD + NN;       // xsum | nsum | msum | cnt
  float*    accbuf  = (float*)   carve((size_t)accN*4);
  float*    xsum = accbuf;
  float*    nsum = accbuf + NN*3;
  float*    msum = accbuf + NN*6;
  float*    cntb = accbuf + NN*6 + (size_t)NN*DD;

  // ---- pack all weights into WMMA-B fragment order (f16, zero-padded K) ----
  #define PACK(SRC, DST, SK, DK) pack_b_kernel<<<((DK)*DD + 255)/256, 256, 0, stream>>>((SRC), (DST), (SK), (DK))
  PACK(ch_W1,   W1p,     CIN,  CINP);
  PACK(ch_W2,   W2p,     DD,   DD);
  PACK(pos_W2,  posW2p,  DD,   DD);
  PACK(sh_W,    shWp,    DD,   DD);
  PACK(phiu_W1, phiuW1p, DD,   DD);
  PACK(phix_W1, phixW1p, DD,   DD);
  PACK(phin_W1, phinW1p, DD,   DD);
  PACK(Wq,      Wqp,     4*DD, 4*DD);
  PACK(Wk,      Wkp,     4*DD, 4*DD);
  PACK(Wv,      Wvp,     4*DD, 4*DD);
  PACK(Wg,      Wgp,     4*DD, 4*DD);
  PACK(out_W,   outWp,   256,  256);
  PACK(phih_W1, phihW1p, 192,  192);
  PACK(phih_W2, phihW2p, DD,   DD);
  #undef PACK

  // ---- edge features / chem / pos / z / phi ----
  edge_kernel<<<(NE+15)/16, 128, 0, stream>>>(
      h, coord, nvecs, edge_attr, node_attr, edges,
      ch_b1, ch_b2, pos_W1, pos_b1, pos_b2, sh_b, att_W, att_b,
      phiu_b1, phiu_W2, phiu_b2, phix_b1, phix_W2, phix_b2,
      phin_b1, phin_W2, phin_b2,
      W1p, W2p, posW2p, shWp, phiuW1p, phixW1p, phinW1p,
      zf16, pupx, pupn, cdiffb);

  // ---- per-head attention ----
  for (int hd = 0; hd < 4; ++hd){
    qkvg_kernel<<<(NE+15)/16, 128, 0, stream>>>(
        zf16, Wqp, Wkp, Wvp, Wgp, Wb, bg, hd, qh, kh, vh, gh, bbuf);
    attn_kernel<<<(NE+3)/4, 128, 0, stream>>>(qh, kh, vh, gh, bbuf, klist, ho[hd]);
  }

  // ---- merge heads ----
  m_kernel<<<(NE+15)/16, 128, 0, stream>>>(ho[0], ho[1], ho[2], ho[3], outWp, out_b, mbuf);

  // ---- segment means ----
  zero_kernel<<<(accN+255)/256, 256, 0, stream>>>(accbuf, accN);
  scatter_kernel<<<((size_t)NE*DD + 255)/256, 256, 0, stream>>>(
      mbuf, pupx, pupn, cdiffb, nvecs, edges, xsum, nsum, msum, cntb);

  // ---- node update ----
  node_kernel<<<(NN+15)/16, 128, 0, stream>>>(
      h, node_attr, coord, nvecs, init_coord, init_nvecs,
      xsum, nsum, msum, cntb, phihW1p, phih_b1, phihW2p, phih_b2, out);
}
